// DeepHodgeLayer_60060822667823
// MI455X (gfx1250) — matrix operand order repaired and checked by reading, not compiled, not run on hardware
//
#include <hip/hip_runtime.h>
#include <hip/hip_bf16.h>
#include <math.h>

// ---------------------------------------------------------------------------
// DeepHodgeLayer for MI455X (gfx1250, wave32, WMMA f32<-f16 16x16x32)
//
// Dense GEMMs -> v_wmma_f32_16x16x32_f16 (f16 operands, f32 accum).
// Boundary-operator einsums (B1/B2, 2-3 nnz per column) -> unrolled sparse
// VALU kernels with compile-time edge enumeration (K=16 fixed).
// L1 (E x E Hodge operator) built per (b,s) in LDS with ds_add_f32 atomics,
// then consumed by a batched WMMA GEMM against transposed-padded E_in.
// ---------------------------------------------------------------------------

typedef _Float16 h16;
typedef __attribute__((ext_vector_type(16))) _Float16 v16h;
typedef __attribute__((ext_vector_type(8)))  _Float16 v8h;
typedef __attribute__((ext_vector_type(8)))  float    v8f;

#define BATCH   256
#define KP      16
#define DM      256
#define NS      3
#define EDGES   120
#define TRIS    560
#define TAUC    1e-4f

// ---------------------------------------------------------------------------
// WMMA fragment loaders (per ISA 7.12.2 layouts, wave32)
// A (16x32 f16): lane L<16 -> row L, K = {0..7, 16..23}; L>=16 -> K+8
// B (32x16 f16): lane L -> col L%16, K = {0..7, 16..23}; L>=16 -> K+8 (same
//                as A) when memory holds W^T (N x K row-major).
// ---------------------------------------------------------------------------
__device__ __forceinline__ v16h load_a_frag(const h16* p) {
  v8h lo = *(const v8h*)p;
  v8h hi = *(const v8h*)(p + 16);
  v16h r;
#pragma unroll
  for (int t = 0; t < 8; t++) { r[t] = lo[t]; r[t + 8] = hi[t]; }
  return r;
}

__device__ __forceinline__ v16h load_b_frag(const h16* p) {
  v8h lo = *(const v8h*)p;
  v8h hi = *(const v8h*)(p + 16);
  v16h r;
#pragma unroll
  for (int t = 0; t < 8; t++) { r[t] = lo[t]; r[t + 8] = hi[t]; }
  return r;
}

// Wave computes a 32(M) x 64(N) tile; A row-major (M x K), WT row-major (N x K)
__device__ __forceinline__ void wmma_tile_32x64(const h16* A, const h16* WT,
                                                int K, int m0, int n0,
                                                v8f acc[2][4]) {
  const int lane  = threadIdx.x & 31;
  const int arow  = lane & 15;
  const int khalf = (lane >> 4) * 8;
  const int bcol  = lane & 15;
  const int bk    = (lane >> 4) * 8;
  for (int k0 = 0; k0 < K; k0 += 32) {
    v16h a0 = load_a_frag(A + (size_t)(m0 + arow)      * K + k0 + khalf);
    v16h a1 = load_a_frag(A + (size_t)(m0 + 16 + arow) * K + k0 + khalf);
#pragma unroll
    for (int j = 0; j < 4; j++) {
      v16h b = load_b_frag(WT + (size_t)(n0 + j * 16 + bcol) * K + k0 + bk);
      acc[0][j] = __builtin_amdgcn_wmma_f32_16x16x32_f16(
          false, a0, false, b, (short)0, acc[0][j], false, false);
      acc[1][j] = __builtin_amdgcn_wmma_f32_16x16x32_f16(
          false, a1, false, b, (short)0, acc[1][j], false, false);
    }
  }
}

// ---------------------------------------------------------------------------
// Generic GEMM with fused epilogues
// ---------------------------------------------------------------------------
#define MODE_V0   0   // f32 out, scatter to (b,s,k,d) layout, +bias
#define MODE_EIN  1   // f16 out, scatter to transposed padded (bs,d,e), +bias
#define MODE_ATTN 2   // f32 out = acc + bias + residual(extra)
#define MODE_FFN1 3   // f16 out = gelu(acc + bias)
#define MODE_FFN2 4   // f32 out = acc + bias + x2(extra)

__global__ __launch_bounds__(128) void gemm_kernel(
    const h16* __restrict__ A, const h16* __restrict__ WT,
    const float* __restrict__ bias, int M, int N, int K, int mode,
    float* __restrict__ outF, h16* __restrict__ outH,
    const float* __restrict__ extra) {
  const int wave = threadIdx.x >> 5;
  const int m0 = blockIdx.y * 128 + wave * 32;
  const int n0 = blockIdx.x * 64;
  v8f acc[2][4] = {};
  wmma_tile_32x64(A, WT, K, m0, n0, acc);

  const int lane  = threadIdx.x & 31;
  const int rbase = (lane >> 4) * 8;
  const int cofs  = lane & 15;
#pragma unroll
  for (int i = 0; i < 2; i++) {
#pragma unroll
    for (int j = 0; j < 4; j++) {
#pragma unroll
      for (int t = 0; t < 8; t++) {
        const int r = m0 + i * 16 + rbase + t;
        const int c = n0 + j * 16 + cofs;
        float v = acc[i][j][t] + bias[c];
        if (mode == MODE_V0) {
          int b = r >> 4, k = r & 15, s = c >> 8, d = c & 255;
          outF[(((size_t)b * NS + s) * KP + k) * DM + d] = v;
        } else if (mode == MODE_EIN) {
          int b = r / EDGES, e = r % EDGES, s = c >> 8, d = c & 255;
          outH[(((size_t)b * NS + s) * DM + d) * 128 + e] = (h16)v;
        } else if (mode == MODE_ATTN) {
          v += extra[(size_t)r * DM + c];
          outF[(size_t)r * DM + c] = v;
        } else if (mode == MODE_FFN1) {
          float g = 0.5f * v * (1.0f + erff(v * 0.70710678118654752440f));
          outH[(size_t)r * N + c] = (h16)g;
        } else { // MODE_FFN2
          v += extra[(size_t)r * DM + c];
          outF[(size_t)r * DM + c] = v;
        }
      }
    }
  }
}

// Batched M1 = L1 @ E_in : 768 GEMMs of (128x128)@(128x256)
__global__ __launch_bounds__(128) void gemm_m1_kernel(
    const h16* __restrict__ L1g, const h16* __restrict__ EinT,
    h16* __restrict__ M1) {
  const int bs = blockIdx.z;
  const h16* A  = L1g + (size_t)bs * 128 * 128;
  const h16* WT = EinT + (size_t)bs * DM * 128;
  const int wave = threadIdx.x >> 5;
  const int m0 = wave * 32;
  const int n0 = blockIdx.x * 64;
  v8f acc[2][4] = {};
  wmma_tile_32x64(A, WT, 128, m0, n0, acc);
  const int lane  = threadIdx.x & 31;
  const int rbase = (lane >> 4) * 8;
  const int cofs  = lane & 15;
#pragma unroll
  for (int i = 0; i < 2; i++)
#pragma unroll
    for (int j = 0; j < 4; j++)
#pragma unroll
      for (int t = 0; t < 8; t++) {
        int r = m0 + i * 16 + rbase + t;
        int c = n0 + j * 16 + cofs;
        M1[((size_t)bs * 128 + r) * DM + c] = (h16)acc[i][j][t];
      }
}

// ---------------------------------------------------------------------------
// Weight convert + transpose: f32 (K x N) -> f16 (N x K)
// ---------------------------------------------------------------------------
__global__ void convt_kernel(const float* __restrict__ w, h16* __restrict__ wt,
                             int K, int N) {
  int idx = blockIdx.x * 256 + threadIdx.x;
  if (idx < K * N) {
    int k = idx / N, n = idx % N;
    wt[(size_t)n * K + k] = (h16)w[idx];
  }
}

// ---------------------------------------------------------------------------
// LayerNorm (+ optional fused geometry projection P = xn @ geom_w + geom_b)
// ---------------------------------------------------------------------------
__global__ __launch_bounds__(256) void ln_kernel(
    const float* __restrict__ x, const float* __restrict__ g,
    const float* __restrict__ bt, h16* __restrict__ out16,
    float* __restrict__ P, const float* __restrict__ gw,
    const float* __restrict__ gb) {
  __shared__ float red[256];
  __shared__ float xs[256];
  const int row = blockIdx.x, tid = threadIdx.x;
  float v = x[(size_t)row * DM + tid];
  red[tid] = v; __syncthreads();
  for (int s = 128; s > 0; s >>= 1) {
    if (tid < s) red[tid] += red[tid + s];
    __syncthreads();
  }
  float mean = red[0] * (1.0f / 256.0f);
  __syncthreads();
  float dv = v - mean;
  red[tid] = dv * dv; __syncthreads();
  for (int s = 128; s > 0; s >>= 1) {
    if (tid < s) red[tid] += red[tid + s];
    __syncthreads();
  }
  float var = red[0] * (1.0f / 256.0f);
  float xn = dv * rsqrtf(var + 1e-5f) * g[tid] + bt[tid];
  out16[(size_t)row * DM + tid] = (h16)xn;
  if (P != nullptr) {
    xs[tid] = xn; __syncthreads();
    if (tid < KP) {
      float s = gb[tid];
      for (int c = 0; c < DM; c++) s += xs[c] * gw[c * KP + tid];
      P[(size_t)row * KP + tid] = s;
    }
  }
}

// E_in_raw[b,e,:] = xn[b,e_i]+xn[b,e_j]  (abs_B1^T einsum, 2 nnz/col)
__global__ void einraw_kernel(const h16* __restrict__ xn16,
                              const int* __restrict__ ei,
                              const int* __restrict__ ej,
                              h16* __restrict__ out) {
  int b = blockIdx.y, e = blockIdx.x, d = threadIdx.x;
  int i = ei[e], j = ej[e];
  float v = (float)xn16[((size_t)b * KP + i) * DM + d] +
            (float)xn16[((size_t)b * KP + j) * DM + d];
  out[((size_t)b * EDGES + e) * DM + d] = (h16)v;
}

// Pairwise distances -> per-scale edge weights W1, triangle weights W2
__global__ __launch_bounds__(256) void d2w_kernel(
    const float* __restrict__ P, const float* __restrict__ mask,
    const float* __restrict__ log_scales, const int* __restrict__ ei,
    const int* __restrict__ ej, const int* __restrict__ tij,
    const int* __restrict__ tjk, const int* __restrict__ tik,
    float* __restrict__ W1, float* __restrict__ W2) {
  __shared__ float Pl[KP * KP];
  __shared__ float W1s[NS * EDGES];
  __shared__ float ml[KP];
  __shared__ float denom[NS];
  const int b = blockIdx.x, tid = threadIdx.x;
  Pl[tid] = P[(size_t)b * KP * KP + tid];
  if (tid < KP) ml[tid] = mask[b * KP + tid];
  if (tid < NS) {
    float sc = expf(log_scales[tid]);
    denom[tid] = 2.0f * sc * sc + 1e-8f;
  }
  __syncthreads();
  if (tid < EDGES) {
    int i = ei[tid], j = ej[tid];
    float m = ml[i] * ml[j];
    float d2 = 0.0f;
#pragma unroll
    for (int c = 0; c < KP; c++) {
      float df = Pl[i * KP + c] - Pl[j * KP + c];
      d2 += df * df;
    }
    d2 *= m * m;
#pragma unroll
    for (int s = 0; s < NS; s++) {
      float w = expf(-d2 / denom[s]) * m;
      W1s[s * EDGES + tid] = w;
      W1[((size_t)b * NS + s) * EDGES + tid] = w;
    }
  }
  __syncthreads();
  for (int idx = tid; idx < NS * TRIS; idx += 256) {
    int s = idx / TRIS, t = idx % TRIS;
    float w = W1s[s * EDGES + tij[t]] * W1s[s * EDGES + tjk[t]] *
              W1s[s * EDGES + tik[t]];
    W2[((size_t)b * NS + s) * TRIS + t] = w;
  }
}

// Zero padded columns e=120..127 of transposed E_in
__global__ void einpad_kernel(h16* __restrict__ eint) {
  size_t row = (size_t)blockIdx.x * 256 + threadIdx.x; // bs*256+d rows
  v8h z = {};
  *(v8h*)(eint + row * 128 + EDGES) = z;
}

// Build L1[b,s] = B1^T diag(mask) B1 + B2 diag(W2) B2^T + tau*I  (120x120 in
// LDS, written out as zero-padded 128x128 f16)
__global__ __launch_bounds__(256) void l1_build_kernel(
    const float* __restrict__ mask, const int* __restrict__ ei,
    const int* __restrict__ ej, const int* __restrict__ tij,
    const int* __restrict__ tjk, const int* __restrict__ tik,
    const float* __restrict__ W2, h16* __restrict__ L1g) {
  __shared__ float L[EDGES * EDGES];
  __shared__ float ml[KP];
  __shared__ int eis[EDGES], ejs[EDGES];
  const int bs = blockIdx.x, b = bs / NS;
  const int tid = threadIdx.x;
  if (tid < KP) ml[tid] = mask[b * KP + tid];
  if (tid < EDGES) { eis[tid] = ei[tid]; ejs[tid] = ej[tid]; }
  __syncthreads();
  // term_down: L[e,f] = sum_{n in e∩f} mask[n]*s(n,e)*s(n,f); s(i)=-1, s(j)=+1
  for (int idx = tid; idx < EDGES * EDGES; idx += 256) {
    int e = idx / EDGES, f = idx % EDGES;
    int a = eis[e], b2 = ejs[e], c = eis[f], d2 = ejs[f];
    float val = 0.0f;
    if (a == c)   val += ml[a];
    if (a == d2)  val -= ml[a];
    if (b2 == c)  val -= ml[b2];
    if (b2 == d2) val += ml[b2];
    if (e == f)   val += TAUC;
    L[idx] = val;
  }
  __syncthreads();
  // term_up: B2 columns have signs +1(ij), +1(jk), -1(ik)
  for (int t = tid; t < TRIS; t += 256) {
    int e1 = tij[t], e2 = tjk[t], e3 = tik[t];
    float w = W2[(size_t)bs * TRIS + t];
    atomicAdd(&L[e1 * EDGES + e1], w);
    atomicAdd(&L[e2 * EDGES + e2], w);
    atomicAdd(&L[e3 * EDGES + e3], w);
    atomicAdd(&L[e1 * EDGES + e2], w);
    atomicAdd(&L[e2 * EDGES + e1], w);
    atomicAdd(&L[e1 * EDGES + e3], -w);
    atomicAdd(&L[e3 * EDGES + e1], -w);
    atomicAdd(&L[e2 * EDGES + e3], -w);
    atomicAdd(&L[e3 * EDGES + e2], -w);
  }
  __syncthreads();
  for (int idx = tid; idx < 128 * 128; idx += 256) {
    int e = idx >> 7, f = idx & 127;
    float v = (e < EDGES && f < EDGES) ? L[e * EDGES + f] : 0.0f;
    L1g[(size_t)bs * 128 * 128 + idx] = (h16)v;
  }
}

// M0 = (B1 diag(W1) B1^T + tau I) V0 : graph-Laplacian apply, fully unrolled,
// writes node_part (left half of all_updates) in f16
__global__ __launch_bounds__(256) void m0_kernel(
    const float* __restrict__ V0, const float* __restrict__ W1,
    h16* __restrict__ allupd) {
  __shared__ float W1s[EDGES];
  const int bs = blockIdx.x, b = bs / NS, s = bs % NS;
  const int d = threadIdx.x;
  if (d < EDGES) W1s[d] = W1[(size_t)bs * EDGES + d];
  __syncthreads();
  float v[KP], acc[KP];
#pragma unroll
  for (int k = 0; k < KP; k++) {
    v[k] = V0[((size_t)bs * KP + k) * DM + d];
    acc[k] = TAUC * v[k];
  }
#pragma unroll
  for (int i = 0; i < KP; i++) {
#pragma unroll
    for (int j = i + 1; j < KP; j++) {
      const int e = (i * (31 - i)) / 2 + (j - i - 1);
      float y = W1s[e] * (v[j] - v[i]);
      acc[j] += y;
      acc[i] -= y;
    }
  }
#pragma unroll
  for (int k = 0; k < KP; k++)
    allupd[((size_t)(b * KP + k)) * (2 * NS * DM) + s * DM + d] = (h16)acc[k];
}

// M1n[k] = sum_{e: k in e} M1[e] : writes edge_part (right half of all_updates)
__global__ __launch_bounds__(256) void m1n_kernel(const h16* __restrict__ M1,
                                                  h16* __restrict__ allupd) {
  const int bs = blockIdx.x, b = bs / NS, s = bs % NS;
  const int d = threadIdx.x;
  float acc[KP];
#pragma unroll
  for (int k = 0; k < KP; k++) acc[k] = 0.0f;
#pragma unroll
  for (int i = 0; i < KP; i++) {
#pragma unroll
    for (int j = i + 1; j < KP; j++) {
      const int e = (i * (31 - i)) / 2 + (j - i - 1);
      float m = (float)M1[((size_t)bs * 128 + e) * DM + d];
      acc[i] += m;
      acc[j] += m;
    }
  }
#pragma unroll
  for (int k = 0; k < KP; k++)
    allupd[((size_t)(b * KP + k)) * (2 * NS * DM) + NS * DM + s * DM + d] =
        (h16)acc[k];
}

// ---------------------------------------------------------------------------
extern "C" void kernel_launch(void* const* d_in, const int* in_sizes, int n_in,
                              void* d_out, int out_size, void* d_ws,
                              size_t ws_size, hipStream_t stream) {
  const float* x          = (const float*)d_in[0];
  const float* mask       = (const float*)d_in[1];
  const int*   e_i        = (const int*)d_in[5];
  const int*   e_j        = (const int*)d_in[6];
  const int*   t_ij       = (const int*)d_in[7];
  const int*   t_jk       = (const int*)d_in[8];
  const int*   t_ik       = (const int*)d_in[9];
  const float* geom_w     = (const float*)d_in[10];
  const float* geom_b     = (const float*)d_in[11];
  const float* log_scales = (const float*)d_in[12];
  const float* wv0_w      = (const float*)d_in[13];
  const float* wv0_b      = (const float*)d_in[14];
  const float* wein_w     = (const float*)d_in[15];
  const float* wein_b     = (const float*)d_in[16];
  const float* out_w      = (const float*)d_in[17];
  const float* out_b      = (const float*)d_in[18];
  const float* n1_g       = (const float*)d_in[19];
  const float* n1_b       = (const float*)d_in[20];
  const float* n2_g       = (const float*)d_in[21];
  const float* n2_b       = (const float*)d_in[22];
  const float* ffn1_w     = (const float*)d_in[23];
  const float* ffn1_b     = (const float*)d_in[24];
  const float* ffn2_w     = (const float*)d_in[25];
  const float* ffn2_b     = (const float*)d_in[26];

  char* ws = (char*)d_ws;
  size_t o = 0;
  auto alloc = [&](size_t bytes) -> char* {
    char* p = ws + o;
    o += (bytes + 255) & ~(size_t)255;
    return p;
  };
  const int ROWS = BATCH * KP;            // 4096
  const int EROWS = BATCH * EDGES;        // 30720
  const int BS = BATCH * NS;              // 768

  h16*   wv0t   = (h16*)alloc((size_t)768 * 256 * 2);
  h16*   weint  = (h16*)alloc((size_t)768 * 256 * 2);
  h16*   outwt  = (h16*)alloc((size_t)256 * 1536 * 2);
  h16*   ffn1t  = (h16*)alloc((size_t)1024 * 256 * 2);
  h16*   ffn2t  = (h16*)alloc((size_t)256 * 1024 * 2);
  h16*   xn16   = (h16*)alloc((size_t)ROWS * DM * 2);      // reused for LN2
  float* Pbuf   = (float*)alloc((size_t)ROWS * KP * 4);
  float* W1buf  = (float*)alloc((size_t)BS * EDGES * 4);
  float* W2buf  = (float*)alloc((size_t)BS * TRIS * 4);
  float* V0buf  = (float*)alloc((size_t)BS * KP * DM * 4);
  h16*   einraw = (h16*)alloc((size_t)EROWS * DM * 2);
  h16*   eint   = (h16*)alloc((size_t)BS * DM * 128 * 2);
  h16*   l1buf  = (h16*)alloc((size_t)BS * 128 * 128 * 2);
  h16*   m1buf  = (h16*)alloc((size_t)BS * 128 * DM * 2);
  h16*   allupd = (h16*)alloc((size_t)ROWS * 2 * NS * DM * 2);
  float* x2buf  = (float*)alloc((size_t)ROWS * DM * 4);
  h16*   hbuf   = (h16*)alloc((size_t)ROWS * 1024 * 2);

  // 1) weight convert + transpose (f32 KxN -> f16 NxK)
  convt_kernel<<<(256 * 768 + 255) / 256, 256, 0, stream>>>(wv0_w, wv0t, 256, 768);
  convt_kernel<<<(256 * 768 + 255) / 256, 256, 0, stream>>>(wein_w, weint, 256, 768);
  convt_kernel<<<(1536 * 256 + 255) / 256, 256, 0, stream>>>(out_w, outwt, 1536, 256);
  convt_kernel<<<(256 * 1024 + 255) / 256, 256, 0, stream>>>(ffn1_w, ffn1t, 256, 1024);
  convt_kernel<<<(1024 * 256 + 255) / 256, 256, 0, stream>>>(ffn2_w, ffn2t, 1024, 256);

  // 2) LN1 + geometry projection
  ln_kernel<<<ROWS, 256, 0, stream>>>(x, n1_g, n1_b, xn16, Pbuf, geom_w, geom_b);

  // 3) E_in_raw = abs_B1^T xn  (sparse)
  einraw_kernel<<<dim3(EDGES, BATCH), 256, 0, stream>>>(xn16, e_i, e_j, einraw);

  // 4) distances -> W1, W2
  d2w_kernel<<<BATCH, 256, 0, stream>>>(Pbuf, mask, log_scales, e_i, e_j, t_ij,
                                        t_jk, t_ik, W1buf, W2buf);

  // 5) V0 = xn @ wv0  (WMMA)
  gemm_kernel<<<dim3(768 / 64, ROWS / 128), 128, 0, stream>>>(
      xn16, wv0t, wv0_b, ROWS, 768, 256, MODE_V0, V0buf, nullptr, nullptr);

  // 6) E_in = E_in_raw @ wein  (WMMA), stored transposed+padded (bs, d, e:128)
  einpad_kernel<<<BS, 256, 0, stream>>>(eint);
  gemm_kernel<<<dim3(768 / 64, EROWS / 128), 128, 0, stream>>>(
      einraw, weint, wein_b, EROWS, 768, 256, MODE_EIN, nullptr, eint, nullptr);

  // 7) build L1 per (b,s) in LDS
  l1_build_kernel<<<BS, 256, 0, stream>>>(mask, e_i, e_j, t_ij, t_jk, t_ik,
                                          W2buf, l1buf);

  // 8) M1 = L1 @ E_in  (batched WMMA)
  gemm_m1_kernel<<<dim3(DM / 64, 1, BS), 128, 0, stream>>>(l1buf, eint, m1buf);

  // 9) node part M0 and edge part M1n -> all_updates (f16)
  m0_kernel<<<BS, 256, 0, stream>>>(V0buf, W1buf, allupd);
  m1n_kernel<<<BS, 256, 0, stream>>>(m1buf, allupd);

  // 10) attn_out + residual -> x2  (WMMA)
  gemm_kernel<<<dim3(DM / 64, ROWS / 128), 128, 0, stream>>>(
      allupd, outwt, out_b, ROWS, DM, 1536, MODE_ATTN, x2buf, nullptr, x);

  // 11) LN2 (reuse xn16 buffer)
  ln_kernel<<<ROWS, 256, 0, stream>>>(x2buf, n2_g, n2_b, xn16, nullptr,
                                      nullptr, nullptr);

  // 12) FFN1 + exact GELU (WMMA)
  gemm_kernel<<<dim3(1024 / 64, ROWS / 128), 128, 0, stream>>>(
      xn16, ffn1t, ffn1_b, ROWS, 1024, 256, MODE_FFN1, nullptr, hbuf, nullptr);

  // 13) FFN2 + residual(x2) -> d_out (WMMA)
  gemm_kernel<<<dim3(DM / 64, ROWS / 128), 128, 0, stream>>>(
      hbuf, ffn2t, ffn2_b, ROWS, DM, 1024, MODE_FFN2, (float*)d_out, nullptr,
      x2buf);
}
